// InstaGraM_13692355740451
// MI455X (gfx1250) — compile-verified, matrix operand order checked by reading, not run on hardware
//
#include <hip/hip_runtime.h>
#include <math.h>

// ---------------- model constants ----------------
#define BB     8
#define KK     400
#define DD     256
#define NCOL   3200          // BB*KK columns, channel-major activations [C][NCOL]
#define HH     200
#define WW     400
#define HCC    25
#define WCC    50
#define NPIX   (HH*WW)       // 80000
#define NMAP   (BB*NPIX)     // 640000
#define SINK_ITERS 50
#define NLAY   7
#define NT_COL 200           // NCOL/16 column tiles
#define KT     25            // KK/16 tiles
#define MS     13            // ceil(KK/32) K-steps over the 400 keys

#define C_NORM   (-6.6846117276679290f)  // -log(800)
#define C_LOG400 ( 5.9914645471079817f)  // log(400)
#define C_LOG800 ( 6.6846117276679290f)  // log(800)

typedef __attribute__((ext_vector_type(16))) _Float16 v16h;
typedef __attribute__((ext_vector_type(8)))  float    v8f;

// K index inside a 32-wide K step for A/B fragment element e (CDNA5 16-bit WMMA layout):
// lanes 0-15 hold K {0..7,16..23}, lanes 16-31 hold K {8..15,24..31}
__device__ __forceinline__ int ig_kmap(int e, int lane) {
  return (e & 7) + ((e >> 3) << 4) + (((lane >> 4) & 1) << 3);
}

// ======== fragment packing: fp32 -> fp16 fragments in exact WMMA lane order ========
// layout: frag[tile][kstep][lane][16 halves]  (32B per lane -> 2x global_load_b128 to consume)

// A-operand from row-major M x Kc matrix (rows map to output M)
__global__ void ig_pack_rows(const float* __restrict__ src, _Float16* __restrict__ dst,
                             int M, int Kc, int MT, int KS) {
  int t = blockIdx.x * 256 + threadIdx.x;      // (mt*KS + ks)*32 + lane
  if (t >= MT * KS * 32) return;
  int lane = t & 31;
  int ks = (t >> 5) % KS;
  int mt = (t >> 5) / KS;
  int row = mt * 16 + (lane & 15);
  v16h f;
#pragma unroll
  for (int e = 0; e < 16; ++e) {
    int k = ks * 32 + ig_kmap(e, lane);
    float v = (row < M && k < Kc) ? src[(size_t)row * Kc + k] : 0.f;
    f[e] = (_Float16)v;
  }
  *(v16h*)(dst + (size_t)t * 16) = f;
}

// B-operand from row-major Kc x N activations (cols map to output N)
__global__ void ig_pack_cols(const float* __restrict__ src, _Float16* __restrict__ dst,
                             int Kc, int KS, int N) {
  int NT = N / 16;
  int t = blockIdx.x * 256 + threadIdx.x;      // (nt*KS + ks)*32 + lane
  if (t >= NT * KS * 32) return;
  int lane = t & 31;
  int ks = (t >> 5) % KS;
  int nt = (t >> 5) / KS;
  int col = nt * 16 + (lane & 15);
  v16h f;
#pragma unroll
  for (int e = 0; e < 16; ++e) {
    int k = ks * 32 + ig_kmap(e, lane);
    float v = (k < Kc) ? src[(size_t)k * N + col] : 0.f;
    f[e] = (_Float16)v;
  }
  *(v16h*)(dst + (size_t)t * 16) = f;
}

// Q/K per-head fragments: idx = ((bh*2 + ks)*KT + tile)*32 + lane ; channel d = kd*4 + h
__global__ void ig_pack_head(const float* __restrict__ src, _Float16* __restrict__ dst) {
  int t = blockIdx.x * 256 + threadIdx.x;      // total 32*2*25*32 = 51200
  if (t >= 32 * 2 * KT * 32) return;
  int lane = t & 31;
  int tile = (t >> 5) % KT;
  int ks = ((t >> 5) / KT) % 2;
  int bh = (t >> 5) / (2 * KT);
  int b = bh >> 2, h = bh & 3;
  int colbase = b * KK + tile * 16 + (lane & 15);
  v16h f;
#pragma unroll
  for (int e = 0; e < 16; ++e) {
    int kd = ks * 32 + ig_kmap(e, lane);
    f[e] = (_Float16)src[(size_t)(kd * 4 + h) * NCOL + colbase];
  }
  *(v16h*)(dst + (size_t)t * 16) = f;
}

// V fragments (A of message): idx = ((bh*4 + hdt)*MS + ks)*32 + lane ; rows=hd, K=m (pad 400->416)
__global__ void ig_pack_v(const float* __restrict__ V, _Float16* __restrict__ dst) {
  int t = blockIdx.x * 256 + threadIdx.x;      // total 32*4*13*32 = 53248
  if (t >= 32 * 4 * MS * 32) return;
  int lane = t & 31;
  int ks = (t >> 5) % MS;
  int hdt = ((t >> 5) / MS) % 4;
  int bh = (t >> 5) / (4 * MS);
  int b = bh >> 2, h = bh & 3;
  int hd = hdt * 16 + (lane & 15);
  v16h f;
#pragma unroll
  for (int e = 0; e < 16; ++e) {
    int m = ks * 32 + ig_kmap(e, lane);
    float v = (m < KK) ? V[(size_t)(hd * 4 + h) * NCOL + b * KK + m] : 0.f;
    f[e] = (_Float16)v;
  }
  *(v16h*)(dst + (size_t)t * 16) = f;
}

// P^T fragments (B of message): idx = ((bh*MS + ks)*KT + nt)*32 + lane ; element (m, n) = P[n][m]
__global__ void ig_pack_p(const float* __restrict__ P, _Float16* __restrict__ dst) {
  int t = blockIdx.x * 256 + threadIdx.x;      // total 32*13*25*32 = 332800
  if (t >= 32 * MS * KT * 32) return;
  int lane = t & 31;
  int nt = (t >> 5) % KT;
  int ks = ((t >> 5) / KT) % MS;
  int bh = (t >> 5) / (MS * KT);
  int n = nt * 16 + (lane & 15);
  v16h f;
#pragma unroll
  for (int e = 0; e < 16; ++e) {
    int m = ks * 32 + ig_kmap(e, lane);
    float v = (m < KK) ? P[(size_t)bh * KK * KK + (size_t)n * KK + m] : 0.f;
    f[e] = (_Float16)v;
  }
  *(v16h*)(dst + (size_t)t * 16) = f;
}

// ge2 fragments for pairwise: idx = ((b*8 + ks)*KT + tile)*32 + lane (used as both A and B)
__global__ void ig_pack_g(const float* __restrict__ G, _Float16* __restrict__ dst) {
  int t = blockIdx.x * 256 + threadIdx.x;      // total 8*8*25*32 = 51200
  if (t >= 8 * 8 * KT * 32) return;
  int lane = t & 31;
  int tile = (t >> 5) % KT;
  int ks = ((t >> 5) / KT) % 8;
  int b = (t >> 5) / (8 * KT);
  int col = b * KK + tile * 16 + (lane & 15);
  v16h f;
#pragma unroll
  for (int e = 0; e < 16; ++e) {
    int d = ks * 32 + ig_kmap(e, lane);
    f[e] = (_Float16)G[(size_t)d * NCOL + col];
  }
  *(v16h*)(dst + (size_t)t * 16) = f;
}

// ======== WMMA consumers: branch-free K loops, 2x 32B loads + 1 wmma per step ========

__global__ void ig_gemm_packed(const _Float16* __restrict__ Wp, const _Float16* __restrict__ Xp,
                               const float* __restrict__ bias, float* __restrict__ out,
                               int M, int KS, int N) {
  int lane = threadIdx.x & 31, wid = threadIdx.x >> 5;
  int nt = blockIdx.x * 4 + wid;
  int mt = blockIdx.y;
  if (nt * 16 >= N) return;
  const v16h* Ap = (const v16h*)Wp + (size_t)mt * KS * 32 + lane;
  const v16h* Bp = (const v16h*)Xp + (size_t)nt * KS * 32 + lane;
  v8f acc = {};
  for (int ks = 0; ks < KS; ++ks) {
    v16h a = Ap[(size_t)ks * 32];
    v16h b = Bp[(size_t)ks * 32];
    acc = __builtin_amdgcn_wmma_f32_16x16x32_f16(false, a, false, b, (short)0, acc, false, false);
  }
  int li = lane & 15, hi = (lane >> 4) & 1;
  int col = nt * 16 + li;
#pragma unroll
  for (int r = 0; r < 8; ++r) {
    int m = mt * 16 + r + hi * 8;
    if (m < M) out[(size_t)m * N + col] = acc[r] + bias[m];
  }
}

__global__ void ig_attn_scores(const _Float16* __restrict__ Qp, const _Float16* __restrict__ Kp,
                               const float* __restrict__ maskA, float* __restrict__ S) {
  int lane = threadIdx.x & 31;
  int bh = blockIdx.z;                          // b*4 + h
  int b = bh >> 2;
  int tn = blockIdx.y, tm = blockIdx.x;
  const v16h* Ap = (const v16h*)Qp + ((size_t)bh * 2 * KT + tn) * 32 + lane;
  const v16h* Bp = (const v16h*)Kp + ((size_t)bh * 2 * KT + tm) * 32 + lane;
  v8f acc = {};
  acc = __builtin_amdgcn_wmma_f32_16x16x32_f16(false, Ap[0], false, Bp[0], (short)0, acc, false, false);
  acc = __builtin_amdgcn_wmma_f32_16x16x32_f16(false, Ap[KT * 32], false, Bp[KT * 32], (short)0, acc, false, false);
  int li = lane & 15, hi = (lane >> 4) & 1;
  int mcol = tm * 16 + li;
  float km = maskA[b * KK + mcol];
  size_t sb = (size_t)bh * KK * KK;
#pragma unroll
  for (int r = 0; r < 8; ++r) {
    int n = tn * 16 + r + hi * 8;
    float v = acc[r] * 0.125f;                  // 1/sqrt(64)
    if (km == 0.f) v = -1e9f;
    S[sb + (size_t)n * KK + mcol] = v;
  }
}

__global__ void ig_attn_message(const _Float16* __restrict__ Vp, const _Float16* __restrict__ Pp,
                                float* __restrict__ Msg) {
  int lane = threadIdx.x & 31;
  int bh = blockIdx.z; int b = bh >> 2, h = bh & 3;
  int hdt = blockIdx.y;                         // 0..3
  int nt  = blockIdx.x;                         // 0..24
  const v16h* Ap = (const v16h*)Vp + ((size_t)(bh * 4 + hdt) * MS) * 32 + lane;
  const v16h* Bp = (const v16h*)Pp + ((size_t)bh * MS * KT + nt) * 32 + lane;
  v8f acc = {};
#pragma unroll
  for (int ks = 0; ks < MS; ++ks) {
    acc = __builtin_amdgcn_wmma_f32_16x16x32_f16(false, Ap[(size_t)ks * 32], false,
                                                 Bp[(size_t)ks * KT * 32], (short)0, acc, false, false);
  }
  int li = lane & 15, hi = (lane >> 4) & 1;
#pragma unroll
  for (int r = 0; r < 8; ++r) {
    int hd = hdt * 16 + r + hi * 8;
    Msg[(size_t)(hd * 4 + h) * NCOL + b * KK + nt * 16 + li] = acc[r];
  }
}

__global__ void ig_pairwise(const _Float16* __restrict__ Gp, const float* __restrict__ maskA,
                            float* __restrict__ Z0) {
  int lane = threadIdx.x & 31;
  int b = blockIdx.z;
  int tn = blockIdx.y, tm = blockIdx.x;
  const v16h* Ap = (const v16h*)Gp + ((size_t)b * 8 * KT + tn) * 32 + lane;
  const v16h* Bp = (const v16h*)Gp + ((size_t)b * 8 * KT + tm) * 32 + lane;
  v8f acc = {};
#pragma unroll
  for (int ks = 0; ks < 8; ++ks) {
    acc = __builtin_amdgcn_wmma_f32_16x16x32_f16(false, Ap[(size_t)ks * KT * 32], false,
                                                 Bp[(size_t)ks * KT * 32], (short)0, acc, false, false);
  }
  int li = lane & 15, hi = (lane >> 4) & 1;
  int mcol = tm * 16 + li;
  float km = maskA[b * KK + mcol];
  size_t zb = (size_t)b * 401 * 401;
#pragma unroll
  for (int r = 0; r < 8; ++r) {
    int n = tn * 16 + r + hi * 8;
    float v = acc[r] * 0.0625f;                 // 1/sqrt(256)
    if (n == mcol) v = -1e9f;
    if (km == 0.f || maskA[b * KK + n] == 0.f) v = -1e9f;
    Z0[zb + (size_t)n * 401 + mcol] = v;
  }
}

// ---------------- tiny 3x256 classifier GEMM (coalesced VALU) ----------------
__global__ void ig_cls_gemm(const float* __restrict__ Wc, const float* __restrict__ bc,
                            const float* __restrict__ X, float* __restrict__ out) {
  int col = blockIdx.x * 256 + threadIdx.x;
  if (col >= NCOL) return;
  float s0 = bc[0], s1 = bc[1], s2 = bc[2];
  for (int d = 0; d < DD; ++d) {
    float x = X[(size_t)d * NCOL + col];
    s0 += Wc[0 * DD + d] * x;
    s1 += Wc[1 * DD + d] * x;
    s2 += Wc[2 * DD + d] * x;
  }
  out[col] = s0; out[NCOL + col] = s1; out[2 * NCOL + col] = s2;
}

// ---------------- vertex cell softmax -> sparse heatmap scatter ----------------
__global__ void ig_cell_softmax(const float* __restrict__ vert, float* __restrict__ smax) {
  int t = blockIdx.x * 256 + threadIdx.x;
  if (t >= BB * HCC * WCC) return;
  int b = t / (HCC * WCC);
  int rem = t % (HCC * WCC);
  int hc = rem / WCC, wc = rem % WCC;
  float vmax = -INFINITY;
  for (int c = 0; c < 65; ++c)
    vmax = fmaxf(vmax, vert[(((size_t)b * 65 + c) * HCC + hc) * WCC + wc]);
  float denom = 0.f;
  for (int c = 0; c < 65; ++c)
    denom += expf(vert[(((size_t)b * 65 + c) * HCC + hc) * WCC + wc] - vmax);
  float best = -INFINITY; int barg = 0;
  for (int c = 0; c < 64; ++c) {
    float e = expf(vert[(((size_t)b * 65 + c) * HCC + hc) * WCC + wc] - vmax);
    if (e > best) { best = e; barg = c; }
  }
  int row = hc * 8 + (barg >> 3);
  int col = wc * 8 + (barg & 7);
  smax[((size_t)b * HH + row) * WW + col] = best / denom;
}

// ---------------- separable 9x9 max pool (SAME, -inf pad) ----------------
__global__ void ig_mp_h(const float* __restrict__ in, float* __restrict__ out) {
  int t = blockIdx.x * 256 + threadIdx.x;
  if (t >= NMAP) return;
  int c = t % WW;
  float m = -INFINITY;
  for (int d = -4; d <= 4; ++d) {
    int cc = c + d;
    if (cc >= 0 && cc < WW) m = fmaxf(m, in[t - c + cc]);
  }
  out[t] = m;
}
__global__ void ig_mp_v(const float* __restrict__ in, float* __restrict__ out) {
  int t = blockIdx.x * 256 + threadIdx.x;
  if (t >= NMAP) return;
  int c = t % WW, r = (t / WW) % HH, b = t / NPIX;
  float m = -INFINITY;
  for (int d = -4; d <= 4; ++d) {
    int rr = r + d;
    if (rr >= 0 && rr < HH) m = fmaxf(m, in[(size_t)b * NPIX + rr * WW + c]);
  }
  out[t] = m;
}
__global__ void ig_eq_mask(const float* s, const float* mp, float* mm) {
  int t = blockIdx.x * 256 + threadIdx.x; if (t >= NMAP) return;
  mm[t] = (s[t] == mp[t]) ? 1.f : 0.f;
}
__global__ void ig_suppress(const float* s, const float* mpm, float* ss) {
  int t = blockIdx.x * 256 + threadIdx.x; if (t >= NMAP) return;
  ss[t] = (mpm[t] > 0.f) ? 0.f : s[t];
}
__global__ void ig_nms_update(float* mm, const float* mpm, const float* ss, const float* mpss) {
  int t = blockIdx.x * 256 + threadIdx.x; if (t >= NMAP) return;
  bool supp = mpm[t] > 0.f;
  bool nm = (ss[t] == mpss[t]);
  if (nm && !supp) mm[t] = 1.f;
}
__global__ void ig_final_mask(const float* s, const float* mm, float* out) {
  int t = blockIdx.x * 256 + threadIdx.x; if (t >= NMAP) return;
  out[t] = (mm[t] > 0.f) ? s[t] : 0.f;
}

// ---------------- per-batch top-400 (iterative argmax, first-index tiebreak) ----------------
__global__ void ig_topk(float* __restrict__ work, float* __restrict__ vals, int* __restrict__ idxo) {
  int b = blockIdx.x;
  float* w = work + (size_t)b * NPIX;
  __shared__ float sv[256]; __shared__ int si[256];
  int t = threadIdx.x;
  for (int k = 0; k < KK; ++k) {
    float bv = -INFINITY; int bi = 0x7fffffff;
    for (int j = t; j < NPIX; j += 256) {
      float x = w[j];
      if (x > bv || (x == bv && j < bi)) { bv = x; bi = j; }
    }
    sv[t] = bv; si[t] = bi; __syncthreads();
    for (int s = 128; s > 0; s >>= 1) {
      if (t < s) {
        if (sv[t + s] > sv[t] || (sv[t + s] == sv[t] && si[t + s] < si[t])) {
          sv[t] = sv[t + s]; si[t] = si[t + s];
        }
      }
      __syncthreads();
    }
    if (t == 0) {
      vals[b * KK + k] = sv[0];
      idxo[b * KK + k] = si[0];
      w[si[0]] = -INFINITY;
    }
    __syncthreads();
  }
}

__global__ void ig_vert_prep(const float* vals, const int* idxi, float* maskA, int* xs, int* ys,
                             float* svA, float* d_maskf, float* d_vs) {
  int t = blockIdx.x * 256 + threadIdx.x;
  if (t >= NCOL) return;
  float v = vals[t]; int id = idxi[t];
  bool m = v > 0.015f;
  int y = m ? id / WW : 0;
  int x = m ? id % WW : 0;
  maskA[t] = m ? 1.f : 0.f;
  xs[t] = x; ys[t] = y;
  svA[t] = m ? v : 0.f;
  d_maskf[t] = m ? 1.f : 0.f;
  d_vs[2 * t + 0] = (float)x;
  d_vs[2 * t + 1] = (float)y;
}

// ---------------- distance patch gather (channel-major [192][NCOL]) ----------------
__global__ void ig_demb(const float* __restrict__ dist, const int* xs, const int* ys,
                        const float* maskA, float* __restrict__ demb) {
  int t = blockIdx.x * 256 + threadIdx.x;
  if (t >= NCOL * 192) return;
  int j = t % 192, col = t / 192;
  int b = col / KK;
  int c = j / 64, r = (j % 64) / 8, cc = j % 8;
  int hc = ys[col] / 8, wc = xs[col] / 8;
  float dv = dist[(((size_t)b * 3 + c) * HH + hc * 8 + r) * WW + wc * 8 + cc];
  dv = fminf(fmaxf(dv, 0.f), 10.f);
  demb[(size_t)j * NCOL + col] = dv * maskA[col];
}

// ---------------- NeRF positional embedding (channel-major [43][NCOL]) ----------------
__global__ void ig_pos(const int* xs, const int* ys, const float* svA, float* __restrict__ pos) {
  int col = blockIdx.x * 256 + threadIdx.x;
  if (col >= NCOL) return;
  float px = ((float)xs[col] - 200.f) / 280.f;
  float py = ((float)ys[col] - 100.f) / 280.f;
  pos[col] = px;
  pos[(size_t)NCOL + col] = py;
  float f = 1.f;
  for (int i = 0; i < 10; ++i) {
    int base = 2 + 4 * i;
    pos[(size_t)(base + 0) * NCOL + col] = sinf(px * f);
    pos[(size_t)(base + 1) * NCOL + col] = sinf(py * f);
    pos[(size_t)(base + 2) * NCOL + col] = cosf(px * f);
    pos[(size_t)(base + 3) * NCOL + col] = cosf(py * f);
    f *= 2.f;
  }
  pos[(size_t)42 * NCOL + col] = svA[col];
}

// ---------------- batch-norm: stats over all NCOL columns, then apply+relu ----------------
__global__ void ig_bn_stats(const float* __restrict__ x, float* mu, float* rs, int N) {
  int c = blockIdx.x;
  const float* xc = x + (size_t)c * N;
  __shared__ float s1[256], s2[256];
  int t = threadIdx.x;
  float a = 0.f, b = 0.f;
  for (int j = t; j < N; j += 256) { float v = xc[j]; a += v; b += v * v; }
  s1[t] = a; s2[t] = b; __syncthreads();
  for (int s = 128; s > 0; s >>= 1) {
    if (t < s) { s1[t] += s1[t + s]; s2[t] += s2[t + s]; }
    __syncthreads();
  }
  if (t == 0) {
    float m = s1[0] / N;
    float var = s2[0] / N - m * m;
    mu[c] = m; rs[c] = rsqrtf(var + 1e-5f);
  }
}
__global__ void ig_bn_apply(float* __restrict__ x, const float* mu, const float* rs,
                            const float* g, const float* be, int C, int N) {
  int t = blockIdx.x * 256 + threadIdx.x;
  if (t >= C * N) return;
  int c = t / N;
  float v = (x[t] - mu[c]) * rs[c] * g[c] + be[c];
  x[t] = fmaxf(v, 0.f);
}

__global__ void ig_add_inplace(float* a, const float* b, int n) {
  int t = blockIdx.x * 256 + threadIdx.x;
  if (t < n) a[t] += b[t];
}

// ---------------- row softmax over 400 keys (mask folded in as -1e9) ----------------
__global__ void ig_softmax_rows(float* __restrict__ S) {
  int row = blockIdx.x;
  float* p = S + (size_t)row * KK;
  __shared__ float red[128];
  int t = threadIdx.x;
  float mx = -INFINITY;
  for (int j = t; j < KK; j += 128) mx = fmaxf(mx, p[j]);
  red[t] = mx; __syncthreads();
  for (int s = 64; s > 0; s >>= 1) { if (t < s) red[t] = fmaxf(red[t], red[t + s]); __syncthreads(); }
  mx = red[0]; __syncthreads();
  float sm = 0.f;
  for (int j = t; j < KK; j += 128) sm += expf(p[j] - mx);
  red[t] = sm; __syncthreads();
  for (int s = 64; s > 0; s >>= 1) { if (t < s) red[t] += red[t + s]; __syncthreads(); }
  float inv = 1.f / red[0];
  for (int j = t; j < KK; j += 128) p[j] = expf(p[j] - mx) * inv;
}

// ---------------- classifier log-softmax over 3 channels ----------------
__global__ void ig_cls_out(const float* __restrict__ clsb, float* __restrict__ out) {
  int col = blockIdx.x * 256 + threadIdx.x;
  if (col >= NCOL) return;
  float a = clsb[col], b1 = clsb[NCOL + col], c = clsb[2 * NCOL + col];
  float mx = fmaxf(a, fmaxf(b1, c));
  float lse = mx + logf(expf(a - mx) + expf(b1 - mx) + expf(c - mx));
  int b = col / KK, n = col % KK;
  out[b * 1200 + 0 * KK + n] = a - lse;
  out[b * 1200 + 1 * KK + n] = b1 - lse;
  out[b * 1200 + 2 * KK + n] = c - lse;
}

// ---------------- Sinkhorn ----------------
__global__ void ig_fill_bins(float* __restrict__ Z0, const float* __restrict__ alpha) {
  int t = blockIdx.x * 256 + threadIdx.x;
  if (t >= BB * 401) return;
  int b = t / 401, i = t % 401;
  float a = alpha[0];
  size_t zb = (size_t)b * 401 * 401;
  Z0[zb + (size_t)i * 401 + 400] = a;
  Z0[zb + (size_t)400 * 401 + i] = a;
}
__global__ void ig_sink_u(const float* __restrict__ Z0, const float* __restrict__ v,
                          float* __restrict__ u) {
  int row = blockIdx.x;             // b*401 + i
  int b = row / 401, i = row % 401;
  const float* z = Z0 + (size_t)b * 401 * 401 + (size_t)i * 401;
  const float* vb = v + b * 401;
  __shared__ float red[128];
  int t = threadIdx.x;
  float mx = -INFINITY;
  for (int j = t; j < 401; j += 128) mx = fmaxf(mx, z[j] + vb[j]);
  red[t] = mx; __syncthreads();
  for (int s = 64; s > 0; s >>= 1) { if (t < s) red[t] = fmaxf(red[t], red[t + s]); __syncthreads(); }
  mx = red[0]; __syncthreads();
  float sm = 0.f;
  for (int j = t; j < 401; j += 128) sm += expf(z[j] + vb[j] - mx);
  red[t] = sm; __syncthreads();
  for (int s = 64; s > 0; s >>= 1) { if (t < s) red[t] += red[t + s]; __syncthreads(); }
  if (t == 0) {
    float lse = mx + logf(red[0]);
    float lm = (i < 400) ? C_NORM : (C_LOG400 + C_NORM);
    u[row] = lm - lse;
  }
}
__global__ void ig_sink_v(const float* __restrict__ Z0, const float* __restrict__ u,
                          float* __restrict__ v) {
  int rc = blockIdx.x;              // b*401 + j
  int b = rc / 401, j = rc % 401;
  const float* z = Z0 + (size_t)b * 401 * 401 + j;
  const float* ub = u + b * 401;
  __shared__ float red[128];
  int t = threadIdx.x;
  float mx = -INFINITY;
  for (int i = t; i < 401; i += 128) mx = fmaxf(mx, z[(size_t)i * 401] + ub[i]);
  red[t] = mx; __syncthreads();
  for (int s = 64; s > 0; s >>= 1) { if (t < s) red[t] = fmaxf(red[t], red[t + s]); __syncthreads(); }
  mx = red[0]; __syncthreads();
  float sm = 0.f;
  for (int i = t; i < 401; i += 128) sm += expf(z[(size_t)i * 401] + ub[i] - mx);
  red[t] = sm; __syncthreads();
  for (int s = 64; s > 0; s >>= 1) { if (t < s) red[t] += red[t + s]; __syncthreads(); }
  if (t == 0) {
    float lse = mx + logf(red[0]);
    float ln = (j < 400) ? C_NORM : (C_LOG400 + C_NORM);
    v[rc] = ln - lse;
  }
}
__global__ void ig_matches_out(const float* __restrict__ Z0, const float* __restrict__ u,
                               const float* __restrict__ v, float* __restrict__ out) {
  int t = blockIdx.x * 256 + threadIdx.x;
  if (t >= BB * 401 * 401) return;
  int b = t / (401 * 401);
  int rem = t % (401 * 401);
  int i = rem / 401, j = rem % 401;
  out[t] = Z0[t] + u[b * 401 + i] + v[b * 401 + j] + C_LOG800;
}

// ================= host orchestration =================
static inline int cdiv(int a, int b) { return (a + b - 1) / b; }

extern "C" void kernel_launch(void* const* d_in, const int* in_sizes, int n_in,
                              void* d_out, int out_size, void* d_ws, size_t ws_size,
                              hipStream_t stream) {
  (void)in_sizes; (void)out_size; (void)ws_size;
  if (n_in < 136) return;   // expects jax tree-flattened (alphabetical) param order

  // --- input indices (setup_inputs dict order, params leaves alphabetical) ---
  const int IN_DIST = 1, IN_VERT = 2, IN_INST = 3, IN_DIR = 4;
  const int IN_BIN = 5, IN_CLSB = 6, IN_CLSW = 7;
  const int IN_DTENC = 8;    // l0..l2: {b,be,g,w}, l3: {b,w}  -> 14 entries
  const int IN_GNN = 22;     // 7 layers x 14: bk,bm,bq,bv, mlp0{b,be,g,w}, mlp1{b,w}, wk,wm,wq,wv
  const int IN_PROJB = 120, IN_PROJW = 121, IN_VENC = 122;
  auto F = [&](int i) { return (const float*)d_in[i]; };

  // --- workspace carve-up (all offsets 32B aligned) ---
  float* wsf = (float*)d_ws;
  size_t off = 0;
  auto A = [&](size_t n) { float* r = wsf + off; off += (n + 7) & ~(size_t)7; return r; };
  float* smax = A(NMAP);   float* tmpb = A(NMAP);
  float* t1   = A(NMAP);   float* t2   = A(NMAP);
  float* t3   = A(NMAP);   float* mmb  = A(NMAP);
  float* work = A(NMAP);
  float* vals = A(NCOL);   int* idxb = (int*)A(NCOL);
  float* maskA = A(NCOL);  int* xs = (int*)A(NCOL); int* ys = (int*)A(NCOL);
  float* svA = A(NCOL);
  float* mu = A(512);      float* rs = A(512);
  float* pos  = A((size_t)43 * NCOL);
  float* demb = A((size_t)192 * NCOL);
  float* actA = A((size_t)512 * NCOL);
  float* actB = A((size_t)512 * NCOL);
  float* catb = A((size_t)512 * NCOL);
  float* ge   = A((size_t)DD * NCOL);
  float* geD  = A((size_t)DD * NCOL);
  float* bufQ = A((size_t)DD * NCOL);
  float* bufK = A((size_t)DD * NCOL);
  float* bufV = A((size_t)DD * NCOL);
  float* bufM = A((size_t)DD * NCOL);
  float* Sb   = A((size_t)BB * 4 * KK * KK);
  float* clsb = A((size_t)3 * NCOL);
  float* Z0   = A((size_t)BB * 401 * 401);
  float* uB   = A((size_t)BB * 401);
  float* vB   = A((size_t)BB * 401);
  // packed fp16 fragment buffers (counts in halves, alloc in float units = halves/2)
  _Float16* Wh = (_Float16*)A(32 * 16 * 512 / 2);          // up to 512x512 weights
  _Float16* Xh = (_Float16*)A((size_t)NT_COL * 16 * 512 / 2); // up to Kc=512 activations
  _Float16* Qp = (_Float16*)A((size_t)32 * 2 * KT * 512 / 2);
  _Float16* Kp = (_Float16*)A((size_t)32 * 2 * KT * 512 / 2);
  _Float16* Vp = (_Float16*)A((size_t)32 * 4 * MS * 512 / 2);
  _Float16* Pp = (_Float16*)A((size_t)32 * MS * KT * 512 / 2);
  _Float16* Gp = (_Float16*)A((size_t)8 * 8 * KT * 512 / 2);

  // --- output offsets (flat concat in return order) ---
  float* out = (float*)d_out;
  const size_t OFF_LOGCLS = 0;
  const size_t OFF_DIST  = 9600;
  const size_t OFF_VERT  = OFF_DIST + (size_t)BB * 3 * HH * WW;       // 1929600
  const size_t OFF_INST  = OFF_VERT + (size_t)BB * 65 * HCC * WCC;    // 2579600
  const size_t OFF_DIR   = OFF_INST + (size_t)BB * HH * WW;           // 3219600
  const size_t OFF_MATCH = OFF_DIR + (size_t)BB * 37 * HCC * WCC;     // 3589600
  const size_t OFF_VS    = OFF_MATCH + (size_t)BB * 401 * 401;        // 4876008
  const size_t OFF_MASKF = OFF_VS + (size_t)BB * KK * 2;              // 4882408

  // --- pass-through outputs ---
  hipMemcpyAsync(out + OFF_DIST, d_in[IN_DIST], (size_t)BB*3*HH*WW*4, hipMemcpyDeviceToDevice, stream);
  hipMemcpyAsync(out + OFF_VERT, d_in[IN_VERT], (size_t)BB*65*HCC*WCC*4, hipMemcpyDeviceToDevice, stream);
  hipMemcpyAsync(out + OFF_INST, d_in[IN_INST], (size_t)BB*HH*WW*4, hipMemcpyDeviceToDevice, stream);
  hipMemcpyAsync(out + OFF_DIR,  d_in[IN_DIR],  (size_t)BB*37*HCC*WCC*4, hipMemcpyDeviceToDevice, stream);

  const int EB = cdiv(NMAP, 256);

  // --- heatmap + NMS ---
  hipMemsetAsync(smax, 0, (size_t)NMAP * 4, stream);
  ig_cell_softmax<<<cdiv(BB*HCC*WCC, 256), 256, 0, stream>>>(F(IN_VERT), smax);
  ig_mp_h<<<EB,256,0,stream>>>(smax, tmpb);  ig_mp_v<<<EB,256,0,stream>>>(tmpb, t1);
  ig_eq_mask<<<EB,256,0,stream>>>(smax, t1, mmb);
  for (int rnd = 0; rnd < 2; ++rnd) {
    ig_mp_h<<<EB,256,0,stream>>>(mmb, tmpb);  ig_mp_v<<<EB,256,0,stream>>>(tmpb, t1);
    ig_suppress<<<EB,256,0,stream>>>(smax, t1, t2);
    ig_mp_h<<<EB,256,0,stream>>>(t2, tmpb);   ig_mp_v<<<EB,256,0,stream>>>(tmpb, t3);
    ig_nms_update<<<EB,256,0,stream>>>(mmb, t1, t2, t3);
  }
  ig_final_mask<<<EB,256,0,stream>>>(smax, mmb, work);

  // --- top-K + vertex prep ---
  ig_topk<<<BB, 256, 0, stream>>>(work, vals, idxb);
  ig_vert_prep<<<cdiv(NCOL,256), 256, 0, stream>>>(vals, idxb, maskA, xs, ys, svA,
                                                   out + OFF_MASKF, out + OFF_VS);

  // --- embeddings ---
  ig_demb<<<cdiv(NCOL*192,256), 256, 0, stream>>>(F(IN_DIST), xs, ys, maskA, demb);
  ig_pos<<<cdiv(NCOL,256), 256, 0, stream>>>(xs, ys, svA, pos);

  auto gemm = [&](const float* Wp_, const float* bp, const float* Xp_, float* Yp, int M, int Kc) {
    int KS = cdiv(Kc, 32), MT = cdiv(M, 16);
    ig_pack_rows<<<cdiv(MT*KS*32, 256), 256, 0, stream>>>(Wp_, Wh, M, Kc, MT, KS);
    ig_pack_cols<<<cdiv(NT_COL*KS*32, 256), 256, 0, stream>>>(Xp_, Xh, Kc, KS, NCOL);
    ig_gemm_packed<<<dim3(NT_COL/4, MT), 128, 0, stream>>>(Wh, Xh, bp, Yp, M, KS, NCOL);
  };
  auto bn = [&](float* Xp_, const float* gp, const float* bep, int C) {
    ig_bn_stats<<<C, 256, 0, stream>>>(Xp_, mu, rs, NCOL);
    ig_bn_apply<<<cdiv(C*NCOL, 256), 256, 0, stream>>>(Xp_, mu, rs, gp, bep, C, NCOL);
  };
  auto encoder = [&](int base, const float* inb, int Cin, float* dst) {
    int chans[5] = {Cin, 64, 128, 256, 256};
    const float* cur = inb;
    for (int i = 0; i < 4; ++i) {
      const float* Wp_ = F(base + (i < 3 ? 4*i + 3 : 13));
      const float* bp  = F(base + (i < 3 ? 4*i + 0 : 12));
      float* o = (i == 3) ? dst : ((i & 1) ? actB : actA);
      gemm(Wp_, bp, cur, o, chans[i+1], chans[i]);
      if (i < 3) bn(o, F(base + 4*i + 2), F(base + 4*i + 1), chans[i+1]);
      cur = o;
    }
  };

  encoder(IN_VENC,  pos,  43,  ge);
  encoder(IN_DTENC, demb, 192, geD);
  ig_add_inplace<<<cdiv(DD*NCOL,256), 256, 0, stream>>>(ge, geD, DD*NCOL);

  // --- 7 GNN layers ---
  for (int L = 0; L < NLAY; ++L) {
    int gb = IN_GNN + L * 14;
    const float *bk = F(gb+0), *bm = F(gb+1), *bq = F(gb+2), *bv = F(gb+3);
    const float *m0b = F(gb+4), *m0be = F(gb+5), *m0g = F(gb+6), *m0w = F(gb+7);
    const float *m1b = F(gb+8), *m1w = F(gb+9);
    const float *wk = F(gb+10), *wm = F(gb+11), *wq = F(gb+12), *wv = F(gb+13);

    gemm(wq, bq, ge, bufQ, DD, DD);
    gemm(wk, bk, ge, bufK, DD, DD);
    gemm(wv, bv, ge, bufV, DD, DD);
    ig_pack_head<<<cdiv(32*2*KT*32,256), 256, 0, stream>>>(bufQ, Qp);
    ig_pack_head<<<cdiv(32*2*KT*32,256), 256, 0, stream>>>(bufK, Kp);
    ig_attn_scores<<<dim3(KT,KT,32), 32, 0, stream>>>(Qp, Kp, maskA, Sb);
    ig_softmax_rows<<<32*KK, 128, 0, stream>>>(Sb);
    ig_pack_v<<<cdiv(32*4*MS*32,256), 256, 0, stream>>>(bufV, Vp);
    ig_pack_p<<<cdiv(32*MS*KT*32,256), 256, 0, stream>>>(Sb, Pp);
    ig_attn_message<<<dim3(KT,4,32), 32, 0, stream>>>(Vp, Pp, bufM);
    gemm(wm, bm, bufM, bufQ, DD, DD);                 // bufQ = merged message
    hipMemcpyAsync(catb, ge, (size_t)DD*NCOL*4, hipMemcpyDeviceToDevice, stream);
    hipMemcpyAsync(catb + (size_t)DD*NCOL, bufQ, (size_t)DD*NCOL*4, hipMemcpyDeviceToDevice, stream);
    gemm(m0w, m0b, catb, actA, 512, 512);
    bn(actA, m0g, m0be, 512);
    gemm(m1w, m1b, actA, actB, DD, 512);
    ig_add_inplace<<<cdiv(DD*NCOL,256), 256, 0, stream>>>(ge, actB, DD*NCOL);  // residual
  }

  // --- projection, classifier, matching ---
  gemm(F(IN_PROJW), F(IN_PROJB), ge, bufQ, DD, DD);   // bufQ = ge2
  ig_cls_gemm<<<cdiv(NCOL,256), 256, 0, stream>>>(F(IN_CLSW), F(IN_CLSB), bufQ, clsb);
  ig_cls_out<<<cdiv(NCOL,256), 256, 0, stream>>>(clsb, out + OFF_LOGCLS);

  ig_pack_g<<<cdiv(8*8*KT*32,256), 256, 0, stream>>>(bufQ, Gp);
  ig_pairwise<<<dim3(KT,KT,BB), 32, 0, stream>>>(Gp, maskA, Z0);
  ig_fill_bins<<<cdiv(BB*401,256), 256, 0, stream>>>(Z0, F(IN_BIN));
  hipMemsetAsync(uB, 0, (size_t)BB*401*4, stream);
  hipMemsetAsync(vB, 0, (size_t)BB*401*4, stream);
  for (int it = 0; it < SINK_ITERS; ++it) {
    ig_sink_u<<<BB*401, 128, 0, stream>>>(Z0, vB, uB);
    ig_sink_v<<<BB*401, 128, 0, stream>>>(Z0, uB, vB);
  }
  ig_matches_out<<<cdiv(BB*401*401,256), 256, 0, stream>>>(Z0, uB, vB, out + OFF_MATCH);
}
